// MLA_1692217114925
// MI455X (gfx1250) — matrix operand order repaired and checked by reading, not compiled, not run on hardware
//
#include <hip/hip_runtime.h>

// ---------------------------------------------------------------------------
// MLA (DeepSeek-style) forward for MI455X / gfx1250, bf16 WMMA everywhere.
//
// Roofline: ~0.73 TFLOP total, ~300 MB HBM traffic -> compute bound on the
// matrix cores. All GEMM-shaped work is routed through
// v_wmma_f32_16x16x32_bf16 (f32 accumulate), attention is a flash-style
// online-softmax kernel operating in the 512-dim KV latent space.
// KV tiles are staged global->LDS with GLOBAL_LOAD_ASYNC_TO_LDS_B128
// (ASYNCcnt-tracked CDNA5 async copy), bypassing VGPRs.
// ---------------------------------------------------------------------------

typedef __attribute__((ext_vector_type(16))) __bf16   v16bf;
typedef __attribute__((ext_vector_type(8)))  float    v8f;
typedef __attribute__((ext_vector_type(4)))  unsigned int u32x4;
typedef unsigned short u16;

#define DD      2048
#define HH      16
#define QLR_    1536
#define KVLR_   512
#define DNOPE_  128
#define DROPE_  64
#define DV_     128
#define QKD_    192
#define BB_     2
#define SS_     2048
#define MTOK    (BB_*SS_)          // 4096 token rows
#define EPS_    1.1920929e-07f
#define SCALE_  0.07216878364870322f   // 192^-0.5

union Frag {
    v16bf v;
    u16   u[16];
    u32x4 q[2];
};

static __device__ inline u16 f2bf(float f) {
    unsigned x = __float_as_uint(f);
    unsigned r = x + 0x7FFFu + ((x >> 16) & 1u);   // round-to-nearest-even
    return (u16)(r >> 16);
}

// A-matrix 16x32 bf16 fragment: lane L holds row (L&15);
// halves 0..15 -> K = 16*g..16*g+15 (g=L>>4), contiguous (matches load_b16).
static __device__ inline void load_a16(Frag& f, const u16* p, int g) {
    f.q[0] = *(const u32x4*)(p + 16 * g);
    f.q[1] = *(const u32x4*)(p + 16 * g + 8);
}
// B-matrix 32x16 bf16 fragment: lane L holds column (L&15);
// lanes 0-15 hold K=0..15, lanes 16-31 hold K=16..31 (contiguous).
static __device__ inline void load_b16(Frag& f, const u16* p, int g) {
    f.q[0] = *(const u32x4*)(p + 16 * g);
    f.q[1] = *(const u32x4*)(p + 16 * g + 8);
}

static __device__ inline v8f wmma_bf16(v16bf a, v16bf b, v8f c) {
    return __builtin_amdgcn_wmma_f32_16x16x32_bf16(
        false, a, false, b, (short)0, c, false, false);
}

// CDNA5 async global->LDS copy, 16 bytes per lane (ASYNCcnt-tracked).
// vdst = VGPR holding LDS byte address, vaddr = 64-bit global address.
static __device__ inline void async_ld_b128(unsigned lds_addr, const void* gaddr) {
    asm volatile("global_load_async_to_lds_b128 %0, %1, off"
                 :: "v"(lds_addr), "v"(gaddr) : "memory");
}
static __device__ inline void wait_async0() {
#if __has_builtin(__builtin_amdgcn_s_wait_asynccnt)
    __builtin_amdgcn_s_wait_asynccnt(0);
#else
    asm volatile("s_wait_asynccnt 0" ::: "memory");
#endif
}
// Flat shared address -> 32-bit LDS byte offset (ISA 10.2: the low 32 bits of the flat address).
static __device__ inline unsigned lds_off(const void* p) {
    return (unsigned)(unsigned long long)p;
}

#define V8F_ZERO ((v8f){0.f,0.f,0.f,0.f,0.f,0.f,0.f,0.f})

// ---------------------------------------------------------------------------
// Generic GEMM: C[M,N] = A[M,K] * W[N,K]^T   (all f32 in global, bf16 WMMA)
// Block = 128 threads (4 waves), tile = 64x64, K-step = 32.
// Batched via gridDim.z with element strides (used for per-head o-up).
// ---------------------------------------------------------------------------
__global__ __launch_bounds__(128) void gemm_bf16_nt(
    const float* __restrict__ A, long long aBatch, int lda,
    const float* __restrict__ W, long long wBatch, int ldw,
    float* __restrict__ C, long long cBatch, int ldc, int K)
{
    __shared__ __attribute__((aligned(16))) u16 Al[64][40];
    __shared__ __attribute__((aligned(16))) u16 Wl[64][40];

    const int tid  = threadIdx.x;
    const int lane = tid & 31, wv = tid >> 5;
    const int ln   = lane & 15, g  = lane >> 4;
    const int n0   = blockIdx.x * 64;
    const int m0   = blockIdx.y * 64;
    const float* Ab = A + (long long)blockIdx.z * aBatch;
    const float* Wb = W + (long long)blockIdx.z * wBatch;
    float*       Cb = C + (long long)blockIdx.z * cBatch;

    v8f acc[4];
#pragma unroll
    for (int nt = 0; nt < 4; ++nt) acc[nt] = V8F_ZERO;

    for (int kb = 0; kb < K; kb += 32) {
#pragma unroll
        for (int i = 0; i < 16; ++i) {       // 2048 elems each of A and W
            int idx = tid + 128 * i;
            int r = idx >> 5, kk = idx & 31;
            Al[r][kk] = f2bf(Ab[(size_t)(m0 + r) * lda + kb + kk]);
            Wl[r][kk] = f2bf(Wb[(size_t)(n0 + r) * ldw + kb + kk]);
        }
        __syncthreads();

        Frag a;
        load_a16(a, &Al[16 * wv + ln][0], g);
#pragma unroll
        for (int nt = 0; nt < 4; ++nt) {
            Frag b;
            load_b16(b, &Wl[16 * nt + ln][0], g);
            acc[nt] = wmma_bf16(a.v, b.v, acc[nt]);
        }
        __syncthreads();
    }

#pragma unroll
    for (int nt = 0; nt < 4; ++nt)
#pragma unroll
        for (int v = 0; v < 8; ++v)
            Cb[(size_t)(m0 + 16 * wv + v + 8 * g) * ldc + n0 + 16 * nt + ln] =
                acc[nt][v];
}

// ---------------------------------------------------------------------------
// RMS norm (f32 -> f32), one block per row.
// ---------------------------------------------------------------------------
__global__ __launch_bounds__(256) void rms_f32(
    const float* __restrict__ X, const float* __restrict__ Wt,
    float* __restrict__ Y, int width)
{
    __shared__ float red[256];
    const int row = blockIdx.x, tid = threadIdx.x;
    const float* x = X + (size_t)row * width;
    float s = 0.f;
    for (int j = tid; j < width; j += 256) { float v = x[j]; s += v * v; }
    red[tid] = s; __syncthreads();
    for (int o = 128; o > 0; o >>= 1) {
        if (tid < o) red[tid] += red[tid + o];
        __syncthreads();
    }
    float sc = rsqrtf(red[0] / (float)width + EPS_);
    for (int j = tid; j < width; j += 256)
        Y[(size_t)row * width + j] = x[j] * sc * Wt[j];
}

// ---------------------------------------------------------------------------
// kv path: rms-norm latent (512 of 576) -> bf16 kv; RoPE the 64-dim k_pe.
// ---------------------------------------------------------------------------
__global__ __launch_bounds__(256) void kv_prep(
    const float* __restrict__ kvfull, const float* __restrict__ kvw,
    const float* __restrict__ cosT, const float* __restrict__ sinT,
    u16* __restrict__ kvb, u16* __restrict__ kpeb)
{
    __shared__ float red[256];
    const int row = blockIdx.x, tid = threadIdx.x;
    const float* x = kvfull + (size_t)row * 576;
    float s = 0.f;
#pragma unroll
    for (int i = 0; i < 2; ++i) { float v = x[tid + 256 * i]; s += v * v; }
    red[tid] = s; __syncthreads();
    for (int o = 128; o > 0; o >>= 1) {
        if (tid < o) red[tid] += red[tid + o];
        __syncthreads();
    }
    float sc = rsqrtf(red[0] / 512.f + EPS_);
#pragma unroll
    for (int i = 0; i < 2; ++i) {
        int j = tid + 256 * i;
        kvb[(size_t)row * 512 + j] = f2bf(x[j] * sc * kvw[j]);
    }
    if (tid < 32) {
        int sidx = row & (SS_ - 1);
        float c = cosT[sidx * 32 + tid], sn = sinT[sidx * 32 + tid];
        float xr = x[512 + 2 * tid], xi = x[512 + 2 * tid + 1];
        kpeb[(size_t)row * 64 + 2 * tid]     = f2bf(xr * c - xi * sn);
        kpeb[(size_t)row * 64 + 2 * tid + 1] = f2bf(xr * sn + xi * c);
    }
}

// ---------------------------------------------------------------------------
// q path post-processing: split q into q_nope (bf16) and RoPE'd q_pe (bf16).
// q layout per row: H heads x (128 nope + 64 rope).
// ---------------------------------------------------------------------------
__global__ __launch_bounds__(256) void q_post(
    const float* __restrict__ Q,
    const float* __restrict__ cosT, const float* __restrict__ sinT,
    u16* __restrict__ qnope, u16* __restrict__ qpeb)
{
    const int row = blockIdx.x, tid = threadIdx.x;
    const float* q = Q + (size_t)row * 3072;
#pragma unroll
    for (int i = 0; i < 8; ++i) {           // 16 heads * 128 = 2048
        int idx = tid + 256 * i;
        int hh = idx >> 7, d = idx & 127;
        qnope[((size_t)row * HH + hh) * 128 + d] = f2bf(q[hh * 192 + d]);
    }
    int sidx = row & (SS_ - 1);
#pragma unroll
    for (int i = 0; i < 2; ++i) {           // 16 heads * 32 pairs = 512
        int idx = tid + 256 * i;
        int hh = idx >> 5, j = idx & 31;
        float c = cosT[sidx * 32 + j], sn = sinT[sidx * 32 + j];
        float xr = q[hh * 192 + 128 + 2 * j];
        float xi = q[hh * 192 + 128 + 2 * j + 1];
        qpeb[((size_t)row * HH + hh) * 64 + 2 * j]     = f2bf(xr * c - xi * sn);
        qpeb[((size_t)row * HH + hh) * 64 + 2 * j + 1] = f2bf(xr * sn + xi * c);
    }
}

__global__ void cvt_bf16(const float* __restrict__ X, u16* __restrict__ Y, int n)
{
    int i = blockIdx.x * blockDim.x + threadIdx.x;
    int stride = gridDim.x * blockDim.x;
    for (; i < n; i += stride) Y[i] = f2bf(X[i]);
}

// ---------------------------------------------------------------------------
// Flash-MLA attention. Grid (S/16, H, B); block 256 = 8 waves.
// Wave w owns latent slice [64w, 64w+64). Waves 0-1 also handle the 64-dim
// RoPE score term (32 dims each). Query tile = 16 rows, KV tile = 32 tokens.
// ---------------------------------------------------------------------------
__global__ __launch_bounds__(256) void attn_kernel(
    const u16* __restrict__ qnope, const u16* __restrict__ qpe,
    const u16* __restrict__ kvl,   const u16* __restrict__ kpe,
    const u16* __restrict__ wkvb,  float* __restrict__ olat)
{
    __shared__ __attribute__((aligned(16))) u16 q_abs_l[16][520];
    __shared__ __attribute__((aligned(16))) u16 kv_l[32][520];
    __shared__ __attribute__((aligned(16))) u16 kpe_l[32][64];
    __shared__ float sc_l[2][16][16];
    __shared__ __attribute__((aligned(16))) u16 p_l[16][32];
    __shared__ float alpha_l[16], m_l[16], l_l[16];

    const int tid  = threadIdx.x;
    const int lane = tid & 31, wv = tid >> 5;
    const int ln   = lane & 15, g  = lane >> 4;
    const int q0 = blockIdx.x * 16;
    const int h  = blockIdx.y;
    const int bb = blockIdx.z;

    // ---- pre-phase: q_abs[16 x 512] = q_nope(16x128) @ wkv_b_K[h](128x512)
    Frag afr[4];
    {
        const u16* qn = qnope + ((size_t)((bb * SS_ + q0 + ln) * HH + h)) * 128;
#pragma unroll
        for (int ks = 0; ks < 4; ++ks) load_a16(afr[ks], qn + 32 * ks, g);
    }
#pragma unroll
    for (int nt = 0; nt < 4; ++nt) {
        v8f c = V8F_ZERO;
#pragma unroll
        for (int ks = 0; ks < 4; ++ks) {
            Frag b;
#pragma unroll
            for (int i = 0; i < 16; ++i)
                b.u[i] = wkvb[(size_t)(h * 256 + 32 * ks + 16 * g + i) * 512 +
                              64 * wv + 16 * nt + ln];
            c = wmma_bf16(afr[ks].v, b.v, c);
        }
#pragma unroll
        for (int v = 0; v < 8; ++v)
            q_abs_l[v + 8 * g][64 * wv + 16 * nt + ln] = f2bf(c[v]);
    }

    Frag ape;
    if (wv < 2)
        load_a16(ape, qpe + ((size_t)((bb * SS_ + q0 + ln) * HH + h)) * 64 +
                          32 * wv, g);

    if (tid < 16) { m_l[tid] = -__builtin_inff(); l_l[tid] = 0.f; }
    __syncthreads();

    v8f acc[4];
#pragma unroll
    for (int nt = 0; nt < 4; ++nt) acc[nt] = V8F_ZERO;

    for (int t0 = 0; t0 <= q0; t0 += 32) {
        // (a) zero partial scores + async global->LDS KV tile copy
        (&sc_l[0][0][0])[tid]       = 0.f;
        (&sc_l[0][0][0])[tid + 256] = 0.f;
        {
            const u16* gsrc = kvl + ((size_t)(bb * SS_ + t0)) * 512;
#pragma unroll
            for (int i = 0; i < 8; ++i) {     // 32x512 bf16 = 2048 x b128
                int j = tid + 256 * i;
                int tok = j >> 6, cc = (j & 63) * 8;
                async_ld_b128(lds_off(&kv_l[tok][cc]), gsrc + (size_t)j * 8);
            }
            const u16* psrc = kpe + ((size_t)(bb * SS_ + t0)) * 64;
            int tok = tid >> 3, cc = (tid & 7) * 8;
            async_ld_b128(lds_off(&kpe_l[tok][cc]), psrc + (size_t)tid * 8);
            if (t0 + 32 <= q0)   // hint next tile into cache (global_prefetch)
                __builtin_prefetch(kvl + ((size_t)(bb * SS_ + t0 + 32)) * 512, 0, 0);
        }
        wait_async0();
        __syncthreads();

        // (b) partial scores: this wave's 64-dim latent slice (+ rope)
        Frag qa[2];
        load_a16(qa[0], &q_abs_l[ln][64 * wv], g);
        load_a16(qa[1], &q_abs_l[ln][64 * wv + 32], g);
#pragma unroll
        for (int sub = 0; sub < 2; ++sub) {
            v8f s = V8F_ZERO;
#pragma unroll
            for (int ks = 0; ks < 2; ++ks) {
                Frag bk;
                load_b16(bk, &kv_l[16 * sub + ln][64 * wv + 32 * ks], g);
                s = wmma_bf16(qa[ks].v, bk.v, s);
            }
            if (wv < 2) {
                Frag bp;
                load_b16(bp, &kpe_l[16 * sub + ln][32 * wv], g);
                s = wmma_bf16(ape.v, bp.v, s);
            }
#pragma unroll
            for (int v = 0; v < 8; ++v)
                atomicAdd(&sc_l[sub][v + 8 * g][ln], s[v]);   // ds_add_f32
        }
        __syncthreads();

        // (c) online softmax, one row-leader thread per query row
        if (tid < 16) {
            const int r = tid, qi = q0 + r;
            float mo = m_l[r], mx = mo;
            float sv[32];
#pragma unroll
            for (int c2 = 0; c2 < 32; ++c2) {
                int t = t0 + c2;
                float xx = sc_l[c2 >> 4][r][c2 & 15] * SCALE_;
                if (t > qi) xx = -__builtin_inff();
                sv[c2] = xx;
                mx = fmaxf(mx, xx);
            }
            float al = __expf(mo - mx);
            float ps = 0.f;
#pragma unroll
            for (int c2 = 0; c2 < 32; ++c2) {
                float p = __expf(sv[c2] - mx);
                p_l[r][c2] = f2bf(p);
                ps += p;
            }
            l_l[r] = l_l[r] * al + ps;
            m_l[r] = mx;
            alpha_l[r] = al;
        }
        __syncthreads();

        // (d) o_slice = o_slice * alpha + P(16x32) @ kv_tile(32 x 64-slice)
        float alv[8];
#pragma unroll
        for (int v = 0; v < 8; ++v) alv[v] = alpha_l[v + 8 * g];
        Frag pa;
        load_a16(pa, &p_l[ln][0], g);
#pragma unroll
        for (int nt = 0; nt < 4; ++nt) {
#pragma unroll
            for (int v = 0; v < 8; ++v) acc[nt][v] *= alv[v];
            Frag bv;
#pragma unroll
            for (int i = 0; i < 16; ++i)
                bv.u[i] = kv_l[16 * g + i][64 * wv + 16 * nt + ln];
            acc[nt] = wmma_bf16(pa.v, bv.v, acc[nt]);
        }
        __syncthreads();
    }

    // epilogue: divide by softmax denominator, store f32 o_latent
    float linv[8];
#pragma unroll
    for (int v = 0; v < 8; ++v) linv[v] = 1.f / l_l[v + 8 * g];
#pragma unroll
    for (int nt = 0; nt < 4; ++nt)
#pragma unroll
        for (int v = 0; v < 8; ++v) {
            int row = q0 + v + 8 * g;
            olat[((size_t)((bb * SS_ + row) * HH + h)) * 512 +
                 64 * wv + 16 * nt + ln] = acc[nt][v] * linv[v];
        }
}

// ---------------------------------------------------------------------------
// Host-side orchestration (all launches on `stream`, graph-capture safe).
// ---------------------------------------------------------------------------
extern "C" void kernel_launch(void* const* d_in, const int* in_sizes, int n_in,
                              void* d_out, int out_size, void* d_ws, size_t ws_size,
                              hipStream_t stream)
{
    (void)in_sizes; (void)n_in; (void)out_size; (void)ws_size;
    const float* x         = (const float*)d_in[0];
    const float* wq_a      = (const float*)d_in[1];
    const float* q_norm_w  = (const float*)d_in[2];
    const float* wq_b      = (const float*)d_in[3];
    const float* wkv_a     = (const float*)d_in[4];
    const float* kv_norm_w = (const float*)d_in[5];
    const float* wkv_b     = (const float*)d_in[6];
    const float* wo        = (const float*)d_in[7];
    const float* fcos      = (const float*)d_in[8];
    const float* fsin      = (const float*)d_in[9];
    // d_in[10] (mask) is exactly causal triu(-inf): handled analytically.
    float* out = (float*)d_out;

    char* wp = (char*)d_ws;
    auto alloc = [&](size_t bytes) -> char* {
        char* p = wp;
        wp += (bytes + 255) & ~(size_t)255;
        return p;
    };
    float* q_lat  = (float*)alloc((size_t)MTOK * QLR_ * 4);        // 25 MB
    float* q_latn = (float*)alloc((size_t)MTOK * QLR_ * 4);        // 25 MB
    float* qf     = (float*)alloc((size_t)MTOK * HH * QKD_ * 4);   // 50 MB
    float* kvfull = (float*)alloc((size_t)MTOK * 576 * 4);         //  9 MB
    u16*   kvb    = (u16*)  alloc((size_t)MTOK * 512 * 2);         //  4 MB
    u16*   kpeb   = (u16*)  alloc((size_t)MTOK * 64 * 2);          //  0.5 MB
    u16*   qnope  = (u16*)  alloc((size_t)MTOK * HH * 128 * 2);    // 17 MB
    u16*   qpeb   = (u16*)  alloc((size_t)MTOK * HH * 64 * 2);     //  8 MB
    u16*   wkvbb  = (u16*)  alloc((size_t)HH * 256 * 512 * 2);     //  4 MB
    float* olat   = (float*)alloc((size_t)MTOK * HH * 512 * 4);    //134 MB
    float* ohead  = (float*)alloc((size_t)MTOK * HH * DV_ * 4);    // 34 MB

    // 1) q_lat = x @ wq_a.T                    [4096,1536]
    gemm_bf16_nt<<<dim3(QLR_/64, MTOK/64, 1), 128, 0, stream>>>(
        x, 0, DD, wq_a, 0, DD, q_lat, 0, QLR_, DD);
    // 2) rms-norm
    rms_f32<<<MTOK, 256, 0, stream>>>(q_lat, q_norm_w, q_latn, QLR_);
    // 3) q = q_latn @ wq_b.T                   [4096,3072]
    gemm_bf16_nt<<<dim3((HH*QKD_)/64, MTOK/64, 1), 128, 0, stream>>>(
        q_latn, 0, QLR_, wq_b, 0, QLR_, qf, 0, HH*QKD_, QLR_);
    // 4) kvfull = x @ wkv_a.T                  [4096,576]
    gemm_bf16_nt<<<dim3(576/64, MTOK/64, 1), 128, 0, stream>>>(
        x, 0, DD, wkv_a, 0, DD, kvfull, 0, 576, DD);
    // 5) kv rms-norm + k_pe rope -> bf16
    kv_prep<<<MTOK, 256, 0, stream>>>(kvfull, kv_norm_w, fcos, fsin, kvb, kpeb);
    // 6) split/rope q -> bf16
    q_post<<<MTOK, 256, 0, stream>>>(qf, fcos, fsin, qnope, qpeb);
    // 7) wkv_b -> bf16 (for attention pre-phase)
    cvt_bf16<<<2048, 256, 0, stream>>>(wkv_b, wkvbb, HH * 256 * KVLR_);
    // 8) flash-MLA attention -> o_latent f32   [4096, H*512]
    attn_kernel<<<dim3(SS_/16, HH, BB_), 256, 0, stream>>>(
        qnope, qpeb, kvb, kpeb, wkvbb, olat);
    // 9) per-head o-up: o_head[:,h*128+d] = o_lat[:,h*512+:] @ wkv_b_V[h].T
    gemm_bf16_nt<<<dim3(DV_/64, MTOK/64, HH), 128, 0, stream>>>(
        olat, (long long)KVLR_, HH * KVLR_,
        wkv_b + (size_t)DNOPE_ * KVLR_, (long long)(DNOPE_ + DV_) * KVLR_, KVLR_,
        ohead, (long long)DV_, HH * DV_, KVLR_);
    // 10) out = o_head @ wo.T                  [4096,2048]
    gemm_bf16_nt<<<dim3(DD/64, MTOK/64, 1), 128, 0, stream>>>(
        ohead, 0, HH * DV_, wo, 0, HH * DV_, out, 0, DD, DD);
}
